// glimpse_network_81071802679537
// MI455X (gfx1250) — compile-verified, matrix operand order checked by reading, not run on hardware
//
#include <hip/hip_runtime.h>
#include <hip/hip_bf16.h>

// ---- problem constants (match reference) ----
#define GB   64            // batch
#define GH   512           // image H
#define GW   512           // image W
#define GC   3             // channels
#define GG   64            // base patch size
#define GKS  3             // num scales
#define D_IN (GKS*GG*GG*GC)  // 36864
#define HG_  128
#define HL_  128
#define OUTC (HG_+HL_)     // 256

typedef __attribute__((ext_vector_type(2)))  float  v2f;
typedef __attribute__((ext_vector_type(8)))  float  v8f;
typedef __attribute__((ext_vector_type(16))) __bf16 v16bf;

// -------------------------------------------------------------------------
// Kernel 1: retina / glimpse extraction.
// One thread per phi element: average-pool an f x f block of the source
// image centered on the (truncated) glimpse coordinate. Out-of-image
// pixels contribute zero (== reference's jnp.pad with zeros).
// phi layout: (B, D_IN) row-major, D_IN index = ((s*G+gy)*G+gx)*C + c.
// -------------------------------------------------------------------------
__global__ __launch_bounds__(256)
void glimpse_extract(const float* __restrict__ x,
                     const float* __restrict__ l,
                     float* __restrict__ phi)
{
    int tid = blockIdx.x * blockDim.x + threadIdx.x;
    const int total = GB * D_IN;
    if (tid >= total) return;

    int b = tid / D_IN;
    int r = tid - b * D_IN;
    int c  = r % GC;
    int t  = r / GC;
    int gx = t % GG;  t /= GG;
    int gy = t % GG;
    int s  = t / GG;                 // scale 0..2

    // denormalize location: trunc(0.5*(l+1)*H); values in [0,512) so (int) == floor
    float lx = l[b * 2 + 0];
    float ly = l[b * 2 + 1];
    int cx = (int)(0.5f * (lx + 1.0f) * (float)GH);
    int cy = (int)(0.5f * (ly + 1.0f) * (float)GH);

    int size = GG << s;              // 64,128,256
    int f    = 1  << s;              // pool factor 1,2,4
    int y0 = cy - (size >> 1) + gy * f;
    int x0 = cx - (size >> 1) + gx * f;

    float acc = 0.0f;
    for (int dy = 0; dy < f; ++dy) {
        int y = y0 + dy;
        if ((unsigned)y < (unsigned)GH) {
            const float* row = x + ((size_t)b * GH + (size_t)y) * (GW * GC);
            for (int dx = 0; dx < f; ++dx) {
                int xx = x0 + dx;
                if ((unsigned)xx < (unsigned)GW)
                    acc += row[xx * GC + c];
            }
        }
    }
    phi[(size_t)b * D_IN + r] = acc / (float)(f * f);
}

// -------------------------------------------------------------------------
// Kernel 2: phi(64 x 36864) @ W1(36864 x 128) + b1, ReLU -> out cols [0,128)
// Grid (4,8): one 16x16 output tile per block. 8 waves split K (4608 each),
// each accumulates with WMMA f32 into v8f; cross-wave reduction via LDS.
// -------------------------------------------------------------------------
__global__ __launch_bounds__(256)
void gemm_wmma(const float* __restrict__ phi,
               const float* __restrict__ W1,
               const float* __restrict__ b1,
               float* __restrict__ out)
{
    const int lane = threadIdx.x & 31;
    const int wave = threadIdx.x >> 5;
    const int half = lane >> 4;          // 0: lanes 0-15, 1: lanes 16-31
    const int m16  = lane & 15;

    const int rowA = blockIdx.x * 16 + m16;   // global batch row for A
    const int colB = blockIdx.y * 16 + m16;   // global output col for B

    const float* __restrict__ Arow = phi + (size_t)rowA * D_IN;

    const int KW = D_IN / 8;             // 4608 K per wave
    const int k0 = wave * KW;
    const int k1 = k0 + KW;

    v8f acc = {};

#if __has_builtin(__builtin_amdgcn_wmma_f32_16x16x4_f32)
    // Native f32 matrix path: 16x16x4, A lane holds {K=kb,kb+1} of row m16,
    // B lane holds {W1[kb][colB], W1[kb+1][colB]}, kb = kk + half*2.
    for (int kk = k0; kk < k1; kk += 4) {
        const int kb = kk + half * 2;
        v2f a, b;
        a.x = Arow[kb];
        a.y = Arow[kb + 1];
        b.x = W1[(size_t)kb * HG_ + colB];
        b.y = W1[(size_t)(kb + 1) * HG_ + colB];
        acc = __builtin_amdgcn_wmma_f32_16x16x4_f32(
            false, a, false, b, (short)0, acc, false, false);
    }
#else
    // Fallback: bf16 16x16x32 WMMA (codegen-confirmed builtin).
    // 16-bit A layout: half 0 -> K = kk+{0..7,16..23}, half 1 -> +8.
    for (int kk = k0; kk < k1; kk += 32) {
        v16bf a, b;
#pragma unroll
        for (int h = 0; h < 16; ++h) {
            const int k = kk + ((h < 8) ? h : (h + 8)) + half * 8;
            a[h] = (__bf16)Arow[k];
            b[h] = (__bf16)W1[(size_t)k * HG_ + colB];
        }
        acc = __builtin_amdgcn_wmma_f32_16x16x32_bf16(
            false, a, false, b, (short)0, acc, false, false);
    }
#endif

    // Cross-wave K reduction through LDS (8 waves x 256 accum floats).
    __shared__ float red[8][256];
#pragma unroll
    for (int r = 0; r < 8; ++r)
        red[wave][r * 32 + lane] = acc[r];
    __syncthreads();

    // Thread t owns (vgpr=r, lane=ll) position; sum the 8 wave partials.
    const int t  = threadIdx.x;
    const int rv = t >> 5;
    const int ll = t & 31;
    float s = 0.0f;
#pragma unroll
    for (int w = 0; w < 8; ++w)
        s += red[w][t];

    // C/D layout: VGPR r, lanes 0-15 -> M=r, lanes 16-31 -> M=r+8; N = ll&15.
    const int mm = rv + (ll >> 4) * 8;
    const int nn = ll & 15;
    const int orow = blockIdx.x * 16 + mm;
    const int ocol = blockIdx.y * 16 + nn;
    const float v = s + b1[ocol];
    out[(size_t)orow * OUTC + ocol] = fmaxf(v, 0.0f);
}

// -------------------------------------------------------------------------
// Kernel 3: l(64x2) @ W2(2x128) + b2, ReLU -> out cols [128,256)
// -------------------------------------------------------------------------
__global__ __launch_bounds__(256)
void loc_fc(const float* __restrict__ l,
            const float* __restrict__ W2,
            const float* __restrict__ b2,
            float* __restrict__ out)
{
    int tid = blockIdx.x * blockDim.x + threadIdx.x;
    if (tid >= GB * HL_) return;
    int b = tid / HL_;
    int j = tid - b * HL_;
    float v = l[b * 2 + 0] * W2[j] + l[b * 2 + 1] * W2[HL_ + j] + b2[j];
    out[(size_t)b * OUTC + HG_ + j] = fmaxf(v, 0.0f);
}

extern "C" void kernel_launch(void* const* d_in, const int* in_sizes, int n_in,
                              void* d_out, int out_size, void* d_ws, size_t ws_size,
                              hipStream_t stream)
{
    const float* x  = (const float*)d_in[0];  // (64,512,512,3)
    const float* l  = (const float*)d_in[1];  // (64,2)
    const float* W1 = (const float*)d_in[2];  // (36864,128)
    const float* b1 = (const float*)d_in[3];  // (128,)
    const float* W2 = (const float*)d_in[4];  // (2,128)
    const float* b2 = (const float*)d_in[5];  // (128,)
    float* out = (float*)d_out;               // (64,256)
    float* phi = (float*)d_ws;                // scratch: 64*36864 floats = 9.44 MB

    const int total = GB * D_IN;
    glimpse_extract<<<(total + 255) / 256, 256, 0, stream>>>(x, l, phi);

    dim3 grid(GB / 16, HG_ / 16);             // (4, 8) 16x16 output tiles
    gemm_wmma<<<grid, 256, 0, stream>>>(phi, W1, b1, out);

    loc_fc<<<(GB * HL_ + 255) / 256, 256, 0, stream>>>(l, W2, b2, out);
}
